// CustomNeighborhoodAttention3D_65678639891308
// MI455X (gfx1250) — matrix operand order repaired and checked by reading, not compiled, not run on hardware
//
#include <hip/hip_runtime.h>
#include <stdint.h>

#define DIM 512
#define NHEADS 8
#define HDIM 64
#define KD 3
#define KH 5
#define KW 5
#define DD 6
#define HH 48
#define WWX 96
#define NVOX (DD * HH * WWX)   // 27648
#define QKV_N (3 * DIM)        // 1536

typedef __attribute__((ext_vector_type(16))) __bf16 v16bf;
typedef __attribute__((ext_vector_type(8)))  float  v8f;

// ---------- bf16 helpers (manual, stored as uint16_t) ----------
__device__ __forceinline__ uint16_t f32_to_bf16(float f) {
  union { float f; uint32_t u; } c; c.f = f;
  uint32_t u = c.u;
  uint32_t r = u + 0x7FFFu + ((u >> 16) & 1u);   // round to nearest even
  if ((u & 0x7F800000u) == 0x7F800000u) r = u;   // inf/nan: truncate
  return (uint16_t)(r >> 16);
}
__device__ __forceinline__ float bf16_to_f32(uint32_t h) {
  union { uint32_t u; float f; } c; c.u = h << 16;
  return c.f;
}

// ---------- prep kernels ----------
__global__ void cvt_bf16(const float* __restrict__ src, uint16_t* __restrict__ dst, int n) {
  int i = blockIdx.x * blockDim.x + threadIdx.x;
  if (i < n) dst[i] = f32_to_bf16(src[i]);
}

// src is K x N row-major (fp32); dst is N x K row-major (bf16)
__global__ void transpose_bf16(const float* __restrict__ src, uint16_t* __restrict__ dst,
                               int K, int N) {
  int i = blockIdx.x * blockDim.x + threadIdx.x;
  if (i < K * N) {
    int k = i / N, n = i % N;
    dst[(size_t)n * K + k] = f32_to_bf16(src[i]);
  }
}

// ---------- WMMA GEMM: C[M x N] = A[M x 512] * Bt[N x 512]^T + bias ----------
// Each wave computes a 32(M) x 64(N) register tile: 2 A fragments x 4 B
// fragments -> 8 WMMAs per 12 b128 loads per k-step. 8 waves/block -> 256x64.
template <bool OUT_BF16>
__global__ __launch_bounds__(256)
void gemm_bf16(const uint16_t* __restrict__ A, const uint16_t* __restrict__ Bt,
               const float* __restrict__ bias, void* __restrict__ out, int N) {
  const int K = DIM;  // 512
  const int wave = threadIdx.x >> 5;
  const int lane = threadIdx.x & 31;
  const int half = lane >> 4;
  const int r    = lane & 15;
  const int m0 = blockIdx.y * 256 + wave * 32;
  const int n0 = blockIdx.x * 64;

  union Frag { v16bf v; uint32_t u[8]; };

  const uint16_t* Arow0 = A + (size_t)(m0 + r) * K;
  const uint16_t* Arow1 = A + (size_t)(m0 + 16 + r) * K;
  v8f acc[8] = {};  // [mt*4 + nt]

  for (int k0 = 0; k0 < K; k0 += 32) {
    // A fragment: 16-bit A 16x32 layout (ISA 7.12.2):
    // VGPR v<4 -> K = 2v (+8*half); v>=4 -> K = 16+2(v-4) (+8*half).
    // Per lane this is two contiguous 16B runs -> compiler emits b128 loads.
    Frag a0, a1;
#pragma unroll
    for (int v = 0; v < 8; ++v) {
      int k = k0 + ((v < 4) ? (2 * v) : (16 + 2 * (v - 4))) + 8 * half;
      a0.u[v] = *(const uint32_t*)(Arow0 + k);
      a1.u[v] = *(const uint32_t*)(Arow1 + k);
    }
#pragma unroll
    for (int nt = 0; nt < 4; ++nt) {
      // B fragment: same K layout as A (v<4 -> K = 2v, v>=4 -> K = 16+2(v-4), +8*half)
      // of column (n0 + nt*16 + r); two contiguous 16B runs per lane.
      const uint16_t* Brow = Bt + (size_t)(n0 + nt * 16 + r) * K;
      Frag b;
#pragma unroll
      for (int v = 0; v < 8; ++v) {
        int k = k0 + ((v < 4) ? (2 * v) : (16 + 2 * (v - 4))) + 8 * half;
        b.u[v] = *(const uint32_t*)(Brow + k);
      }
      acc[nt] = __builtin_amdgcn_wmma_f32_16x16x32_bf16(
          false, a0.v, false, b.v, (short)0, acc[nt], false, false);
      acc[4 + nt] = __builtin_amdgcn_wmma_f32_16x16x32_bf16(
          false, a1.v, false, b.v, (short)0, acc[4 + nt], false, false);
    }
  }

  // D layout: lane r = N column within tile, VGPR v -> M = v + 8*half.
#pragma unroll
  for (int mt = 0; mt < 2; ++mt) {
#pragma unroll
    for (int nt = 0; nt < 4; ++nt) {
#pragma unroll
      for (int v = 0; v < 8; ++v) {
        int m = m0 + mt * 16 + v + 8 * half;
        int n = n0 + nt * 16 + r;
        float val = acc[mt * 4 + nt][v] + bias[n];
        if (OUT_BF16) ((uint16_t*)out)[(size_t)m * N + n] = f32_to_bf16(val);
        else          ((float*)out)[(size_t)m * N + n]    = val;
      }
    }
  }
}

// ---------- NA3D attention: one wave32 per (voxel, head), 2 channels/lane ----------
__global__ __launch_bounds__(256)
void na3d_attn(const uint16_t* __restrict__ qkv, uint16_t* __restrict__ out) {
  int gid  = blockIdx.x * blockDim.x + threadIdx.x;
  int lane = gid & 31;
  int wid  = gid >> 5;
  int head = wid & (NHEADS - 1);
  int vox  = wid >> 3;
  if (vox >= NVOX) return;

  int w0 = vox % WWX;
  int h0 = (vox / WWX) % HH;
  int d0 = vox / (WWX * HH);
  int sd = min(max(d0 - KD / 2, 0), DD  - KD);
  int sh = min(max(h0 - KH / 2, 0), HH  - KH);
  int sw = min(max(w0 - KW / 2, 0), WWX - KW);

  const float qscale = 1.0f / 64.0f;  // reference: scale*scale = HEAD_DIM^-1
  size_t qoff = (size_t)vox * QKV_N + head * HDIM + 2 * lane;
  uint32_t qu = *(const uint32_t*)(qkv + qoff);
  float q0 = bf16_to_f32(qu & 0xFFFFu) * qscale;
  float q1 = bf16_to_f32(qu >> 16)     * qscale;

  float mmax = -INFINITY, lsum = 0.0f, a0 = 0.0f, a1 = 0.0f;

  for (int dz = 0; dz < KD; ++dz)
    for (int hy = 0; hy < KH; ++hy)
      for (int wx = 0; wx < KW; ++wx) {
        int nrow = ((sd + dz) * HH + (sh + hy)) * WWX + (sw + wx);
        size_t kb = (size_t)nrow * QKV_N + DIM + head * HDIM + 2 * lane;
        uint32_t ku = *(const uint32_t*)(qkv + kb);
        float s = q0 * bf16_to_f32(ku & 0xFFFFu) + q1 * bf16_to_f32(ku >> 16);
#pragma unroll
        for (int off = 16; off > 0; off >>= 1) s += __shfl_xor(s, off, 32);

        float nm   = fmaxf(mmax, s);
        float corr = __expf(mmax - nm);   // first iter: exp(-inf)=0
        float p    = __expf(s - nm);
        uint32_t vu = *(const uint32_t*)(qkv + kb + DIM);
        lsum = lsum * corr + p;
        a0   = a0 * corr + p * bf16_to_f32(vu & 0xFFFFu);
        a1   = a1 * corr + p * bf16_to_f32(vu >> 16);
        mmax = nm;
      }

  float inv = 1.0f / lsum;
  size_t ob = (size_t)vox * DIM + head * HDIM + 2 * lane;
  uint32_t packed = (uint32_t)f32_to_bf16(a0 * inv) |
                    ((uint32_t)f32_to_bf16(a1 * inv) << 16);
  *(uint32_t*)(out + ob) = packed;
}

// ---------- launch ----------
extern "C" void kernel_launch(void* const* d_in, const int* in_sizes, int n_in,
                              void* d_out, int out_size, void* d_ws, size_t ws_size,
                              hipStream_t stream) {
  const float* x      = (const float*)d_in[0];
  const float* w_qkv  = (const float*)d_in[1];
  const float* b_qkv  = (const float*)d_in[2];
  const float* w_proj = (const float*)d_in[3];
  const float* b_proj = (const float*)d_in[4];

  char* ws = (char*)d_ws;
  size_t off = 0;
  auto alloc = [&](size_t bytes) -> void* {
    void* p = ws + off;
    off += (bytes + 255) & ~(size_t)255;
    return p;
  };
  uint16_t* x_bf    = (uint16_t*)alloc((size_t)NVOX * DIM * 2);    // 28.3 MB
  uint16_t* wqkvT   = (uint16_t*)alloc((size_t)QKV_N * DIM * 2);   // 1.6 MB
  uint16_t* wprojT  = (uint16_t*)alloc((size_t)DIM * DIM * 2);     // 0.5 MB
  uint16_t* qkv_bf  = (uint16_t*)alloc((size_t)NVOX * QKV_N * 2);  // 84.9 MB
  uint16_t* attn_bf = (uint16_t*)alloc((size_t)NVOX * DIM * 2);    // 28.3 MB

  {
    int n = NVOX * DIM;
    cvt_bf16<<<(n + 255) / 256, 256, 0, stream>>>(x, x_bf, n);
  }
  transpose_bf16<<<(DIM * QKV_N + 255) / 256, 256, 0, stream>>>(w_qkv, wqkvT, DIM, QKV_N);
  transpose_bf16<<<(DIM * DIM + 255) / 256, 256, 0, stream>>>(w_proj, wprojT, DIM, DIM);

  // qkv = x @ w_qkv + b_qkv   (bf16 WMMA, fp32 accumulate, bf16 out)
  dim3 g1(QKV_N / 64, NVOX / 256);
  gemm_bf16<true><<<g1, 256, 0, stream>>>(x_bf, wqkvT, b_qkv, qkv_bf, QKV_N);

  // neighborhood attention
  {
    long long nthreads = (long long)NVOX * NHEADS * 32;
    na3d_attn<<<(unsigned)(nthreads / 256), 256, 0, stream>>>(qkv_bf, attn_bf);
  }

  // out = attn @ w_proj + b_proj  (fp32 out straight to d_out)
  dim3 g2(DIM / 64, NVOX / 256);
  gemm_bf16<false><<<g2, 256, 0, stream>>>(attn_bf, wprojT, b_proj, d_out, DIM);
}
